// ConsciousnessTransformer_55087250539152
// MI455X (gfx1250) — matrix operand order repaired and checked by reading, not compiled, not run on hardware
//
#include <hip/hip_runtime.h>
#include <math.h>
#include <stdint.h>

typedef __bf16 bf16;
typedef __attribute__((ext_vector_type(16))) bf16  v16bf;
typedef __attribute__((ext_vector_type(8)))  float v8f;

typedef __attribute__((address_space(3))) bf16 lds_bf16;

// ---------------------------------------------------------------------------
// Elementwise helpers
// ---------------------------------------------------------------------------
__device__ __forceinline__ float gelu_f(float x) {
    return 0.5f * x * (1.0f + erff(x * 0.70710678118654752f));
}

__global__ void cast_bf16_kernel(const float* __restrict__ in, bf16* __restrict__ out, long long n) {
    long long i = (long long)blockIdx.x * blockDim.x + threadIdx.x;
    long long stride = (long long)gridDim.x * blockDim.x;
    for (; i < n; i += stride) out[i] = (bf16)in[i];
}

// ---------------------------------------------------------------------------
// Tiled transpose + f32->bf16 cast:  out[c][r] = (bf16) in[r][c]
// grid = (C/32, R/32, batches), block = 256
// ---------------------------------------------------------------------------
__global__ __launch_bounds__(256)
void transpose_cast_kernel(const float* __restrict__ in, long long sIo, long long sIi, int ldin,
                           bf16* __restrict__ out, long long sOo, long long sOi, int ldout,
                           int innerN) {
    __shared__ bf16 tile[32][33];
    int z  = blockIdx.z;
    int zo = z / innerN, zi = z - zo * innerN;
    const float* ib = in + zo * sIo + zi * sIi;
    bf16*        ob = out + zo * sOo + zi * sOi;
    int c0 = blockIdx.x * 32, r0 = blockIdx.y * 32;
    int tx = threadIdx.x & 31, ty = threadIdx.x >> 5;   // ty: 0..7
    #pragma unroll
    for (int i = 0; i < 4; ++i) {
        int r = r0 + ty + i * 8;
        tile[ty + i * 8][tx] = (bf16)ib[(long long)r * ldin + c0 + tx];
    }
    __syncthreads();
    #pragma unroll
    for (int i = 0; i < 4; ++i) {
        int c = c0 + ty + i * 8;
        ob[(long long)c * ldout + r0 + tx] = tile[tx][ty + i * 8];
    }
}

// ---------------------------------------------------------------------------
// Embedding: h[b,n,:] = tok[x[b,n]] + pos[n] + skill[sid[b]] + emo[eid[b]]
// ---------------------------------------------------------------------------
__global__ __launch_bounds__(256)
void embed_kernel(const int* __restrict__ x, const int* __restrict__ sid, const int* __restrict__ eid,
                  const float* __restrict__ tok, const float* __restrict__ pos,
                  const float* __restrict__ skill, const float* __restrict__ emo,
                  float* __restrict__ h) {
    int rn = blockIdx.x;            // b*512 + n
    int b  = rn >> 9;
    int n  = rn & 511;
    int t  = threadIdx.x;
    long long tokid = x[rn];
    long long sb = sid[b], eb = eid[b];
    float* dst = h + (long long)rn * 512;
    #pragma unroll
    for (int i = 0; i < 2; ++i) {
        int d = t + i * 256;
        dst[d] = tok[tokid * 512 + d] + pos[(long long)n * 512 + d]
               + skill[sb * 512 + d] + emo[eb * 512 + d];
    }
}

// ---------------------------------------------------------------------------
// LayerNorm over D=512, one block per row. Optional f32 and bf16 outputs.
// ---------------------------------------------------------------------------
__global__ __launch_bounds__(256)
void layernorm512_kernel(const float* __restrict__ in, const float* __restrict__ w,
                         const float* __restrict__ bia, float* __restrict__ outf,
                         bf16* __restrict__ outb) {
    long long row = blockIdx.x;
    const float* x = in + row * 512;
    int t = threadIdx.x;
    float x0 = x[t], x1 = x[t + 256];
    __shared__ float red[256];
    red[t] = x0 + x1;
    __syncthreads();
    for (int o = 128; o > 0; o >>= 1) { if (t < o) red[t] += red[t + o]; __syncthreads(); }
    float mean = red[0] * (1.0f / 512.0f);
    __syncthreads();
    float d0 = x0 - mean, d1 = x1 - mean;
    red[t] = d0 * d0 + d1 * d1;
    __syncthreads();
    for (int o = 128; o > 0; o >>= 1) { if (t < o) red[t] += red[t + o]; __syncthreads(); }
    float rstd = rsqrtf(red[0] * (1.0f / 512.0f) + 1e-5f);
    float y0 = d0 * rstd * w[t] + bia[t];
    float y1 = d1 * rstd * w[t + 256] + bia[t + 256];
    if (outf) { outf[row * 512 + t] = y0; outf[row * 512 + t + 256] = y1; }
    if (outb) { outb[row * 512 + t] = (bf16)y0; outb[row * 512 + t + 256] = (bf16)y1; }
}

// ---------------------------------------------------------------------------
// Strided-batched bf16 GEMM, f32 accumulate:  T = alpha * A[M,K] * B[K,N] (+bias +resid)
// then: Cf (f32, optional) and C16 (bf16, optional) receive act(T) (act: 0=none, 1=gelu).
// A row-major [M][K] (lda), B K-major [K][N] (ldb)  -- weights pre-transposed.
// Block = 256 threads (8 waves). Macro tile 128(M) x 64(N), K-step 32.
// Wave w: 32x32 tile: 2x2 WMMA accumulators (v_wmma_f32_16x16x32_bf16).
// Staging is done with GLOBAL_LOAD_ASYNC_TO_LDS_B128 (ASYNCcnt-tracked) into a
// double-buffered LDS tile; one s_wait_asynccnt + one barrier per K-step.
// Requires: M%128==0, N%64==0, K%32==0, 16B-aligned A/B bases & leading dims.
// ---------------------------------------------------------------------------
__global__ __launch_bounds__(256)
void gemm_bf16_kernel(const bf16* __restrict__ A, long long sAo, long long sAi, int lda,
                      const bf16* __restrict__ B, long long sBo, long long sBi, int ldb,
                      const float* __restrict__ bias,
                      const float* __restrict__ resid, long long sRo, long long sRi, int ldr,
                      float* __restrict__ Cf, bf16* __restrict__ C16,
                      long long sCo, long long sCi, int ldc,
                      int K, int innerN, float alpha, int act) {
    __shared__ bf16 la[2][128][40];   // [buf][m][k], 80B rows (16B-aligned)
    __shared__ bf16 lb[2][32][72];    // [buf][k][n], 144B rows (16B-aligned)

    int z  = blockIdx.z;
    int zo = z / innerN, zi = z - zo * innerN;
    const bf16*  Ab  = A + zo * sAo + zi * sAi;
    const bf16*  Bb  = B + zo * sBo + zi * sBi;
    float*       Cb  = Cf  ? (Cf  + zo * sCo + zi * sCi) : nullptr;
    bf16*        Cb16= C16 ? (C16 + zo * sCo + zi * sCi) : nullptr;
    const float* Rb  = resid ? (resid + zo * sRo + zi * sRi) : nullptr;

    int t    = threadIdx.x;
    int lane = t & 31;
    int wave = t >> 5;
    int wm   = (wave & 3) << 5;   // 0,32,64,96
    int wn   = (wave >> 2) << 5;  // 0,32
    long long m0 = (long long)blockIdx.y * 128;
    long long n0 = (long long)blockIdx.x * 64;

    // staging coordinates (all 16B-aligned)
    const int am = t >> 1;               // A row 0..127
    const int ak = (t & 1) << 4;         // A k offset 0/16
    const int bk = t >> 3;               // B k row 0..31
    const int bn = (t & 7) << 3;         // B n offset 0..56 step 8

    // Async-copy one K-step tile (A: 2x16B per thread, B: 1x16B) into buffer `buf`.
    auto stageAsync = [&](int k0, int buf) {
        lds_bf16* pa = (lds_bf16*)&la[buf][am][ak];
        lds_bf16* pb = (lds_bf16*)&lb[buf][bk][bn];
        unsigned long long ga = (unsigned long long)(uintptr_t)
            (Ab + (m0 + am) * (long long)lda + (k0 + ak));
        unsigned long long gb = (unsigned long long)(uintptr_t)
            (Bb + (long long)(k0 + bk) * ldb + (n0 + bn));
        asm volatile("global_load_async_to_lds_b128 %0, %1, off"
                     :: "v"(pa), "v"(ga) : "memory");
        asm volatile("global_load_async_to_lds_b128 %0, %1, off offset:16"
                     :: "v"(pa), "v"(ga) : "memory");
        asm volatile("global_load_async_to_lds_b128 %0, %1, off"
                     :: "v"(pb), "v"(gb) : "memory");
    };

    v8f acc00 = {}, acc01 = {}, acc10 = {}, acc11 = {};

    const int nsteps = K >> 5;
    stageAsync(0, 0);                    // prologue: tile 0 -> buffer 0

    const int ar = wm + (lane & 15);
    const int kb = (lane >> 4) << 3;     // 0 / 8

    for (int s = 0; s < nsteps; ++s) {
        asm volatile("s_wait_asynccnt 0x0" ::: "memory");  // my staged tile landed
        __syncthreads();                                   // everyone's tile landed
        int cur = s & 1;
        if (s + 1 < nsteps) stageAsync((s + 1) << 5, 1 - cur);

        // Fragments from buffer `cur`.
        // A 16x32: lanes 0-15 -> M=lane, K {0..7,16..23}; lanes 16-31 -> M=lane-16, K {8..15,24..31}
        v16bf af0, af1, bf0v, bf1v;
        #pragma unroll
        for (int e = 0; e < 16; ++e) {
            int kk = (e & 7) + ((e >> 3) << 4) + kb;
            af0[e] = la[cur][ar][kk];
            af1[e] = la[cur][ar + 16][kk];
        }
        // B 32x16: lanes -> N=lane&15, K same as A
        #pragma unroll
        for (int e = 0; e < 16; ++e) {
            int kk = (e & 7) + ((e >> 3) << 4) + kb;
            bf0v[e] = lb[cur][kk][wn + (lane & 15)];
            bf1v[e] = lb[cur][kk][wn + 16 + (lane & 15)];
        }

        acc00 = __builtin_amdgcn_wmma_f32_16x16x32_bf16(false, af0, false, bf0v, (short)0, acc00, false, false);
        acc01 = __builtin_amdgcn_wmma_f32_16x16x32_bf16(false, af0, false, bf1v, (short)0, acc01, false, false);
        acc10 = __builtin_amdgcn_wmma_f32_16x16x32_bf16(false, af1, false, bf0v, (short)0, acc10, false, false);
        acc11 = __builtin_amdgcn_wmma_f32_16x16x32_bf16(false, af1, false, bf1v, (short)0, acc11, false, false);
    }

    // Epilogue. C/D layout: VGPR r -> M = r (lanes 0-15) / 8+r (lanes 16-31), N = lane&15.
    int cl = lane & 15;
    long long col0 = n0 + wn + cl;
    long long col1 = col0 + 16;
    long long rowb = m0 + wm + ((lane >> 4) << 3);
    float bv0 = bias ? bias[col0] : 0.0f;
    float bv1 = bias ? bias[col1] : 0.0f;
    #pragma unroll
    for (int r = 0; r < 8; ++r) {
        long long r0i = rowb + r;
        long long r1i = r0i + 16;
        float v00 = alpha * acc00[r] + bv0;
        float v01 = alpha * acc01[r] + bv1;
        float v10 = alpha * acc10[r] + bv0;
        float v11 = alpha * acc11[r] + bv1;
        if (Rb) {
            v00 += Rb[r0i * (long long)ldr + col0];
            v01 += Rb[r0i * (long long)ldr + col1];
            v10 += Rb[r1i * (long long)ldr + col0];
            v11 += Rb[r1i * (long long)ldr + col1];
        }
        if (act == 1) {
            v00 = gelu_f(v00); v01 = gelu_f(v01);
            v10 = gelu_f(v10); v11 = gelu_f(v11);
        }
        if (Cb) {
            Cb[r0i * (long long)ldc + col0] = v00;
            Cb[r0i * (long long)ldc + col1] = v01;
            Cb[r1i * (long long)ldc + col0] = v10;
            Cb[r1i * (long long)ldc + col1] = v11;
        }
        if (Cb16) {
            Cb16[r0i * (long long)ldc + col0] = (bf16)v00;
            Cb16[r0i * (long long)ldc + col1] = (bf16)v01;
            Cb16[r1i * (long long)ldc + col0] = (bf16)v10;
            Cb16[r1i * (long long)ldc + col1] = (bf16)v11;
        }
    }
}

// ---------------------------------------------------------------------------
// Row softmax over 512 columns -> bf16 out. One block per row.
// ---------------------------------------------------------------------------
__global__ __launch_bounds__(256)
void softmax512_kernel(const float* __restrict__ in, bf16* __restrict__ out) {
    long long row = blockIdx.x;
    const float* x = in + row * 512;
    int t = threadIdx.x;
    float a = x[t], b = x[t + 256];
    __shared__ float red[256];
    red[t] = fmaxf(a, b);
    __syncthreads();
    for (int o = 128; o > 0; o >>= 1) { if (t < o) red[t] = fmaxf(red[t], red[t + o]); __syncthreads(); }
    float m = red[0];
    __syncthreads();
    float e0 = expf(a - m), e1 = expf(b - m);
    red[t] = e0 + e1;
    __syncthreads();
    for (int o = 128; o > 0; o >>= 1) { if (t < o) red[t] += red[t + o]; __syncthreads(); }
    float inv = 1.0f / red[0];
    bf16* y = out + row * 512;
    y[t] = (bf16)(e0 * inv);
    y[t + 256] = (bf16)(e1 * inv);
}

// ---------------------------------------------------------------------------
// Mean pool over N=512 tokens. One thread per (b,d).
// ---------------------------------------------------------------------------
__global__ __launch_bounds__(256)
void meanpool_kernel(const float* __restrict__ hf, float* __restrict__ pooled,
                     float* __restrict__ outp) {
    int idx = blockIdx.x * blockDim.x + threadIdx.x;   // 0..16383
    int b = idx >> 9, d = idx & 511;
    const float* p = hf + (long long)b * 512 * 512 + d;
    float s = 0.0f;
    for (int n = 0; n < 512; ++n) s += p[(long long)n * 512];
    s *= (1.0f / 512.0f);
    pooled[idx] = s;
    outp[idx] = s;
}

// ---------------------------------------------------------------------------
// Small dense layer for heads: out[r,j] = act(b[j] + dot(in[r,:], W[j,:]))
// act: 0=none, 1=gelu, 2=tanh, 3=sigmoid. grid = rows, block = 256.
// ---------------------------------------------------------------------------
__global__ __launch_bounds__(256)
void linact_kernel(const float* __restrict__ in, int IN,
                   const float* __restrict__ W, const float* __restrict__ bia,
                   float* __restrict__ out, int OUT, int act) {
    int r = blockIdx.x;
    const float* x = in + (long long)r * IN;
    for (int j = threadIdx.x; j < OUT; j += blockDim.x) {
        const float* wr = W + (long long)j * IN;
        float s = bia[j];
        for (int i = 0; i < IN; ++i) s += x[i] * wr[i];
        if (act == 1)      s = gelu_f(s);
        else if (act == 2) s = tanhf(s);
        else if (act == 3) s = 1.0f / (1.0f + expf(-s));
        out[(long long)r * OUT + j] = s;
    }
}

// Tiny row softmax (rows x cols, cols small). One thread per row.
__global__ void softmax_small_kernel(const float* __restrict__ in, float* __restrict__ out,
                                     int rows, int cols) {
    int r = blockIdx.x * blockDim.x + threadIdx.x;
    if (r >= rows) return;
    const float* x = in + (long long)r * cols;
    float m = x[0];
    for (int c = 1; c < cols; ++c) m = fmaxf(m, x[c]);
    float s = 0.0f;
    for (int c = 0; c < cols; ++c) s += expf(x[c] - m);
    float inv = 1.0f / s;
    for (int c = 0; c < cols; ++c) out[(long long)r * cols + c] = expf(x[c] - m) * inv;
}

// ---------------------------------------------------------------------------
// Host launcher
// ---------------------------------------------------------------------------
extern "C" void kernel_launch(void* const* d_in, const int* in_sizes, int n_in,
                              void* d_out, int out_size, void* d_ws, size_t ws_size,
                              hipStream_t stream) {
    (void)in_sizes; (void)n_in; (void)out_size; (void)ws_size;

    // ---- inputs (setup_inputs dict insertion order, params flattened recursively)
    const int*   xi      = (const int*)d_in[0];
    const int*   sid     = (const int*)d_in[1];
    const int*   eid     = (const int*)d_in[2];
    const float* tok     = (const float*)d_in[3];
    const float* pos     = (const float*)d_in[4];
    const float* skill   = (const float*)d_in[5];
    const float* emo     = (const float*)d_in[6];
    const float* ln1_w   = (const float*)d_in[7];
    const float* ln1_b   = (const float*)d_in[8];
    const float* qkv_w   = (const float*)d_in[9];
    const float* qkv_b   = (const float*)d_in[10];
    const float* proj_w  = (const float*)d_in[11];
    const float* proj_b  = (const float*)d_in[12];
    const float* ln2_w   = (const float*)d_in[13];
    const float* ln2_b   = (const float*)d_in[14];
    const float* fc1_w   = (const float*)d_in[15];
    const float* fc1_b   = (const float*)d_in[16];
    const float* fc2_w   = (const float*)d_in[17];
    const float* fc2_b   = (const float*)d_in[18];
    const float* norm_w  = (const float*)d_in[19];
    const float* norm_b  = (const float*)d_in[20];
    const float* sk1_w   = (const float*)d_in[21];
    const float* sk1_b   = (const float*)d_in[22];
    const float* sk2_w   = (const float*)d_in[23];
    const float* sk2_b   = (const float*)d_in[24];
    const float* em1_w   = (const float*)d_in[25];
    const float* em1_b   = (const float*)d_in[26];
    const float* em2_w   = (const float*)d_in[27];
    const float* em2_b   = (const float*)d_in[28];
    const float* co1_w   = (const float*)d_in[29];
    const float* co1_b   = (const float*)d_in[30];
    const float* co2_w   = (const float*)d_in[31];
    const float* co2_b   = (const float*)d_in[32];
    const float* cr1_w   = (const float*)d_in[33];
    const float* cr1_b   = (const float*)d_in[34];
    const float* cr2_w   = (const float*)d_in[35];
    const float* cr2_b   = (const float*)d_in[36];
    const float* wi1_w   = (const float*)d_in[37];
    const float* wi1_b   = (const float*)d_in[38];
    const float* wi2_w   = (const float*)d_in[39];
    const float* wi2_b   = (const float*)d_in[40];
    const float* im1_w   = (const float*)d_in[41];
    const float* im1_b   = (const float*)d_in[42];
    const float* im2_w   = (const float*)d_in[43];
    const float* im2_b   = (const float*)d_in[44];

    const long long TOK = 16384;                 // B*N
    char* ws = (char*)d_ws;

    // ---- workspace layout (bytes, 256-aligned)
    size_t off = 0;
    auto take = [&](size_t bytes) { size_t o = off; off = (off + bytes + 255) & ~(size_t)255; return o; };
    size_t o_h    = take(TOK * 512 * 4);          // residual stream f32
    size_t o_xnb  = take(TOK * 512 * 2);          // LN output bf16
    size_t o_qkv  = take(TOK * 1536 * 4);         // qkv f32 (K^T transpose source)
    size_t o_qkvb = take(TOK * 1536 * 2);         // qkv bf16 (Q rows, V K-major)
    size_t o_kT   = take(32ll * 8 * 64 * 512 * 2);// K^T bf16 [b,h,64,512]
    size_t o_ob   = take(TOK * 512 * 2);          // attention out bf16
    size_t o_ffb  = take(TOK * 2048 * 2);         // gelu(FC1) bf16
    size_t o_sc   = take(64ll * 512 * 512 * 4);   // scores f32 (8 batches x 8 heads)
    size_t o_pb   = take(64ll * 512 * 512 * 2);   // probs bf16
    size_t o_hf   = take(TOK * 512 * 4);          // final LN f32
    size_t o_pool = take(32 * 512 * 4);           // pooled
    size_t o_hid  = take(32 * 1024 * 4);          // head hidden
    size_t o_etmp = take(32 * 25 * 4);            // emotion logits
    size_t o_wq   = take(12ll * 512 * 1536 * 2);  // bf16 transposed weights [K][N]
    size_t o_wp   = take(12ll * 512 * 512 * 2);
    size_t o_w1   = take(12ll * 512 * 2048 * 2);
    size_t o_w2   = take(12ll * 2048 * 512 * 2);

    float* h    = (float*)(ws + o_h);
    bf16*  xnb  = (bf16*) (ws + o_xnb);
    float* qkvf = (float*)(ws + o_qkv);
    bf16*  qkvb = (bf16*) (ws + o_qkvb);
    bf16*  kT   = (bf16*) (ws + o_kT);
    bf16*  ob   = (bf16*) (ws + o_ob);
    bf16*  ffb  = (bf16*) (ws + o_ffb);
    float* scf  = (float*)(ws + o_sc);
    bf16*  prb  = (bf16*) (ws + o_pb);
    float* hf   = (float*)(ws + o_hf);
    float* pool = (float*)(ws + o_pool);
    float* hid  = (float*)(ws + o_hid);
    float* etmp = (float*)(ws + o_etmp);
    bf16*  wqT  = (bf16*) (ws + o_wq);
    bf16*  wpT  = (bf16*) (ws + o_wp);
    bf16*  w1T  = (bf16*) (ws + o_w1);
    bf16*  w2T  = (bf16*) (ws + o_w2);

    // ---- transpose+cast all block weights once: W[N][K] f32 -> W^T[K][N] bf16
    transpose_cast_kernel<<<dim3(16, 48, 12), 256, 0, stream>>>(
        qkv_w, 1536ll * 512, 0, 512, wqT, 512ll * 1536, 0, 1536, 1);
    transpose_cast_kernel<<<dim3(16, 16, 12), 256, 0, stream>>>(
        proj_w, 512ll * 512, 0, 512, wpT, 512ll * 512, 0, 512, 1);
    transpose_cast_kernel<<<dim3(16, 64, 12), 256, 0, stream>>>(
        fc1_w, 2048ll * 512, 0, 512, w1T, 512ll * 2048, 0, 2048, 1);
    transpose_cast_kernel<<<dim3(64, 16, 12), 256, 0, stream>>>(
        fc2_w, 512ll * 2048, 0, 2048, w2T, 2048ll * 512, 0, 512, 1);

    // ---- embedding
    embed_kernel<<<16384, 256, 0, stream>>>(xi, sid, eid, tok, pos, skill, emo, h);

    // ---- transformer blocks
    for (int l = 0; l < 12; ++l) {
        // LN1 -> bf16
        layernorm512_kernel<<<16384, 256, 0, stream>>>(h, ln1_w + l * 512, ln1_b + l * 512,
                                                       nullptr, xnb);
        // QKV: [16384,512] x [512,1536] + bias  -> f32 (for K^T) AND bf16 (Q/V)
        gemm_bf16_kernel<<<dim3(24, 128, 1), 256, 0, stream>>>(
            xnb, 0, 0, 512,
            wqT + (long long)l * 512 * 1536, 0, 0, 1536,
            qkv_b + (long long)l * 1536,
            nullptr, 0, 0, 0,
            qkvf, qkvb, 0, 0, 1536,
            512, 1, 1.0f, 0);
        // K^T: per (b,h): [512(n),64(d)] (row stride 1536) -> [64][512] bf16
        transpose_cast_kernel<<<dim3(2, 16, 256), 256, 0, stream>>>(
            qkvf + 512, 512ll * 1536, 64, 1536,
            kT, 8ll * 64 * 512, 64ll * 512, 512, 8);

        // Attention in chunks of 8 batches (scores buffer = 8 b x 8 h x 512 x 512)
        for (int b0 = 0; b0 < 32; b0 += 8) {
            long long qoff = (long long)b0 * 512 * 1536;
            // S = (1/8) * Q K^T : per (b,h), M=512, N=512, K=64
            gemm_bf16_kernel<<<dim3(8, 4, 64), 256, 0, stream>>>(
                qkvb + qoff, 512ll * 1536, 64, 1536,
                kT + (long long)b0 * 8 * 64 * 512, 8ll * 64 * 512, 64ll * 512, 512,
                nullptr,
                nullptr, 0, 0, 0,
                scf, nullptr, 8ll * 512 * 512, 512ll * 512, 512,
                64, 8, 0.125f, 0);
            softmax512_kernel<<<32768, 256, 0, stream>>>(scf, prb);
            // O = P V : per (b,h), M=512, N=64, K=512 (V already K-major) -> bf16 direct
            gemm_bf16_kernel<<<dim3(1, 4, 64), 256, 0, stream>>>(
                prb, 8ll * 512 * 512, 512ll * 512, 512,
                qkvb + qoff + 1024, 512ll * 1536, 64, 1536,
                nullptr,
                nullptr, 0, 0, 0,
                nullptr, ob + (long long)b0 * 512 * 512, 512ll * 512, 64, 512,
                512, 8, 1.0f, 0);
        }

        // proj + residual into h
        gemm_bf16_kernel<<<dim3(8, 128, 1), 256, 0, stream>>>(
            ob, 0, 0, 512,
            wpT + (long long)l * 512 * 512, 0, 0, 512,
            proj_b + (long long)l * 512,
            h, 0, 0, 512,
            h, nullptr, 0, 0, 512,
            512, 1, 1.0f, 0);

        // LN2 -> bf16
        layernorm512_kernel<<<16384, 256, 0, stream>>>(h, ln2_w + l * 512, ln2_b + l * 512,
                                                       nullptr, xnb);
        // FC1 + bias + GELU fused -> bf16 direct
        gemm_bf16_kernel<<<dim3(32, 128, 1), 256, 0, stream>>>(
            xnb, 0, 0, 512,
            w1T + (long long)l * 512 * 2048, 0, 0, 2048,
            fc1_b + (long long)l * 2048,
            nullptr, 0, 0, 0,
            nullptr, ffb, 0, 0, 2048,
            512, 1, 1.0f, 1);
        // FC2 + residual into h
        gemm_bf16_kernel<<<dim3(8, 128, 1), 256, 0, stream>>>(
            ffb, 0, 0, 2048,
            w2T + (long long)l * 2048 * 512, 0, 0, 512,
            fc2_b + (long long)l * 512,
            h, 0, 0, 512,
            h, nullptr, 0, 0, 512,
            2048, 1, 1.0f, 0);
    }

    // ---- final LN (f32) + mean pool
    layernorm512_kernel<<<16384, 256, 0, stream>>>(h, norm_w, norm_b, hf, nullptr);

    float* out = (float*)d_out;
    meanpool_kernel<<<64, 256, 0, stream>>>(hf, pool, out + 0);   // pooled -> out[0:16384]

    // ---- heads (tiny; plain VALU)
    linact_kernel<<<32, 256, 0, stream>>>(pool, 512, sk1_w, sk1_b, hid, 1024, 1);
    linact_kernel<<<32, 256, 0, stream>>>(hid, 1024, sk2_w, sk2_b, out + 16384, 50, 3);
    linact_kernel<<<32, 256, 0, stream>>>(pool, 512, em1_w, em1_b, hid, 512, 1);
    linact_kernel<<<32, 256, 0, stream>>>(hid, 512, em2_w, em2_b, etmp, 25, 0);
    softmax_small_kernel<<<1, 32, 0, stream>>>(etmp, out + 17984, 32, 25);
    linact_kernel<<<32, 256, 0, stream>>>(pool, 512, co1_w, co1_b, hid, 512, 1);
    linact_kernel<<<32, 256, 0, stream>>>(hid, 512, co2_w, co2_b, out + 18784, 1, 3);
    linact_kernel<<<32, 256, 0, stream>>>(pool, 512, cr1_w, cr1_b, hid, 1024, 1);
    linact_kernel<<<32, 256, 0, stream>>>(hid, 1024, cr2_w, cr2_b, out + 18816, 512, 0);
    linact_kernel<<<32, 256, 0, stream>>>(pool, 512, wi1_w, wi1_b, hid, 512, 2);
    linact_kernel<<<32, 256, 0, stream>>>(hid, 512, wi2_w, wi2_b, out + 35200, 256, 0);
    linact_kernel<<<32, 256, 0, stream>>>(pool, 512, im1_w, im1_b, hid, 512, 1);
    linact_kernel<<<32, 256, 0, stream>>>(hid, 512, im2_w, im2_b, out + 43392, 512, 0);
}
